// Attention_62826781606241
// MI455X (gfx1250) — matrix operand order repaired and checked by reading, not compiled, not run on hardware
//
#include <hip/hip_runtime.h>
#include <stdint.h>

// ---------------------------------------------------------------------------
// Batched attention (B=16, Tq=128, Tk=2048, D=1024, fp32) for gfx1250.
//
// Fast path (needs 192MB workspace):
//   k1: enc f32 -> Khi/Klo bf16 planes [b][k][d]   (RNE hi + exact-residual lo)
//   k2: enc f32 -> Vt bf16 plane [b][d][k]         (transposed via LDS tile)
//   k3: per (batch, 16-query tile) WG:
//       pass1: S^T = K * Q^T      (3-term split bf16 wmma, all loads contiguous)
//       softmax on S^T (f32, LDS), P^T -> bf16 hi/lo planes in LDS
//       pass2: ctx^T = V^T * P^T  (2-term wmma, contiguous loads)
// Fallback: round-1 inline-conversion kernel (correct, more VALU).
// ---------------------------------------------------------------------------

typedef __attribute__((ext_vector_type(16))) __bf16 bf16x16;
typedef __attribute__((ext_vector_type(8)))  float  f32x8;

union alignas(32) BF16Vec { uint32_t u[8]; uint4 q[2]; bf16x16 v; };

constexpr int Bn  = 16;
constexpr int TKn = 2048;
constexpr int TQn = 128;
constexpr int Dn  = 1024;
constexpr int QT  = 16;
constexpr int NWAVE = 8;

// ---- workspace layout (bytes) ----
constexpr size_t KHI_OFF = 0;                                    // [B][Tk][D] u16
constexpr size_t KLO_OFF = (size_t)Bn * TKn * Dn * 2;            // 64MB
constexpr size_t VT_OFF  = KLO_OFF * 2;                          // [B][D][Tk] u16
constexpr size_t WS_NEED = VT_OFF + (size_t)Bn * Dn * TKn * 2;   // 192MB

// ---- LDS layout for attn_ws (bytes) ----
constexpr int QSTR = 1032;                 // Q plane row stride (u16), banks: 4*ll
constexpr int PSTR = 2056;                 // P plane row stride (u16), banks: 4*ll
constexpr int SSTR = 17;                   // S^T row stride (f32), bank-safe
constexpr int QHI_O = 0;
constexpr int QLO_O = QT * QSTR * 2;                   // 33024
constexpr int PHI_O = 0;                               // P planes overlay Q region
constexpr int PLO_O = QT * PSTR * 2;                   // 65792
constexpr int REGA_SZ = 2 * QT * PSTR * 2;             // 131584
constexpr int S_O   = REGA_SZ;
constexpr int INV_O = S_O + TKn * SSTR * 4;            // +139264 = 270848
constexpr int LDSZ  = INV_O + 64;                      // 270912 (< 320KB/WGP)

// truncating split (in-kernel; residual captured exactly by lo)
__device__ __forceinline__ void splitT(float x0, float x1,
                                       uint32_t& hi, uint32_t& lo) {
  uint32_t u0 = __float_as_uint(x0), u1 = __float_as_uint(x1);
  uint32_t h0 = u0 & 0xFFFF0000u,    h1 = u1 & 0xFFFF0000u;
  hi = (u0 >> 16) | h1;
  float l0 = x0 - __uint_as_float(h0);
  float l1 = x1 - __uint_as_float(h1);
  lo = (__float_as_uint(l0) >> 16) | (__float_as_uint(l1) & 0xFFFF0000u);
}

__device__ __forceinline__ uint32_t bf16rne(float x) {
  uint32_t u = __float_as_uint(x);
  return (u + 0x7FFFu + ((u >> 16) & 1u)) >> 16;
}

__device__ __forceinline__ f32x8 wmma_bf16(bf16x16 a, bf16x16 b, f32x8 c) {
  return __builtin_amdgcn_wmma_f32_16x16x32_bf16(false, a, false, b,
                                                 (short)0, c, false, false);
}

// ---------------------------------------------------------------------------
// k1: elementwise f32 -> (hi, lo) bf16 planes, 8 elements/thread
// ---------------------------------------------------------------------------
__global__ __launch_bounds__(256) void conv_k_planes(
    const float* __restrict__ src, uint16_t* __restrict__ hi,
    uint16_t* __restrict__ lo) {
  size_t i0 = ((size_t)blockIdx.x * 256 + threadIdx.x) * 8;
  float4 a = *(const float4*)(src + i0);
  float4 c = *(const float4*)(src + i0 + 4);
  float xs[8] = {a.x, a.y, a.z, a.w, c.x, c.y, c.z, c.w};
  uint32_t hp[4], lp[4];
#pragma unroll
  for (int j = 0; j < 4; ++j) {
    uint32_t h0 = bf16rne(xs[2 * j]), h1 = bf16rne(xs[2 * j + 1]);
    float l0 = xs[2 * j]     - __uint_as_float(h0 << 16);
    float l1 = xs[2 * j + 1] - __uint_as_float(h1 << 16);
    hp[j] = h0 | (h1 << 16);
    lp[j] = bf16rne(l0) | (bf16rne(l1) << 16);
  }
  *(uint4*)(hi + i0) = *(uint4*)hp;
  *(uint4*)(lo + i0) = *(uint4*)lp;
}

// ---------------------------------------------------------------------------
// k2: transpose f32 [b][k][d] -> bf16 [b][d][k], 64x64 tiles via LDS
// ---------------------------------------------------------------------------
__global__ __launch_bounds__(256) void conv_vt_plane(
    const float* __restrict__ src, uint16_t* __restrict__ vt) {
  __shared__ uint16_t tile[64][68];
  const int b = blockIdx.z, k0 = blockIdx.x * 64, d0 = blockIdx.y * 64;
  const int tr = threadIdx.x >> 4, tc = threadIdx.x & 15;
  const float* sp = src + ((size_t)b * TKn + k0) * Dn + d0;
#pragma unroll
  for (int i = 0; i < 4; ++i) {
    int r = tr + 16 * i;
    float4 f = *(const float4*)(sp + (size_t)r * Dn + tc * 4);
    uint32_t p0 = bf16rne(f.x) | (bf16rne(f.y) << 16);
    uint32_t p1 = bf16rne(f.z) | (bf16rne(f.w) << 16);
    *(uint2*)&tile[r][tc * 4] = make_uint2(p0, p1);
  }
  __syncthreads();
  uint16_t* dst = vt + ((size_t)b * Dn + d0) * TKn + k0;
#pragma unroll
  for (int i = 0; i < 4; ++i) {
    int d = tr + 16 * i;
    uint16_t v0 = tile[tc * 4 + 0][d], v1 = tile[tc * 4 + 1][d];
    uint16_t v2 = tile[tc * 4 + 2][d], v3 = tile[tc * 4 + 3][d];
    uint32_t p0 = (uint32_t)v0 | ((uint32_t)v1 << 16);
    uint32_t p1 = (uint32_t)v2 | ((uint32_t)v3 << 16);
    *(uint2*)(dst + (size_t)d * TKn + tc * 4) = make_uint2(p0, p1);
  }
}

// ---------------------------------------------------------------------------
// k3: attention using precomputed planes (no inline f32->bf16 in hot loops)
// ---------------------------------------------------------------------------
__global__ __launch_bounds__(256) void attn_ws(
    const float* __restrict__ dec, const uint16_t* __restrict__ khi,
    const uint16_t* __restrict__ klo, const uint16_t* __restrict__ vt,
    float* __restrict__ out) {
  extern __shared__ char smem[];
  float* S    = (float*)(smem + S_O);
  float* sInv = (float*)(smem + INV_O);

  const int tid = threadIdx.x, b = blockIdx.y, q0 = blockIdx.x * QT;
  const int w = tid >> 5, lane = tid & 31, ll = lane & 15, half = lane >> 4;

  // ---- stage Q tile as bf16 hi/lo planes [q][d], stride QSTR ----
  {
    const int row = tid >> 4, seg = tid & 15;
    const float* src = dec + ((size_t)b * TQn + q0 + row) * Dn + seg * 64;
    uint32_t* qh = (uint32_t*)(smem + QHI_O);
    uint32_t* ql = (uint32_t*)(smem + QLO_O);
#pragma unroll
    for (int i = 0; i < 16; ++i) {
      float4 f = ((const float4*)src)[i];
      uint32_t h0, l0, h1, l1;
      splitT(f.x, f.y, h0, l0);
      splitT(f.z, f.w, h1, l1);
      int di = row * (QSTR / 2) + seg * 32 + i * 2;
      qh[di] = h0; qh[di + 1] = h1;
      ql[di] = l0; ql[di + 1] = l1;
    }
  }
  __syncthreads();

  // ---- pass 1: S^T = K * Q^T  (A = K planes, B = Q^T from LDS) ----
  {
    const uint16_t* khb = khi + (size_t)b * TKn * Dn;
    const uint16_t* klb = klo + (size_t)b * TKn * Dn;
    const uint16_t* qhP = (const uint16_t*)(smem + QHI_O);
    const uint16_t* qlP = (const uint16_t*)(smem + QLO_O);
    for (int kt = w; kt < TKn / 16; kt += NWAVE) {
      const int k0 = kt * 16;
      f32x8 acc = {0.f, 0.f, 0.f, 0.f, 0.f, 0.f, 0.f, 0.f};
      const uint16_t* ah = khb + (size_t)(k0 + ll) * Dn;
      const uint16_t* al = klb + (size_t)(k0 + ll) * Dn;
#pragma unroll 2
      for (int c = 0; c < Dn / 32; ++c) {
        const int ao = c * 32 + 16 * half;         // A: k-row, 16 consec d
        BF16Vec Ah, Al, Bh, Bl;
        Ah.q[0] = *(const uint4*)(ah + ao);
        Ah.q[1] = *(const uint4*)(ah + ao + 8);
        Al.q[0] = *(const uint4*)(al + ao);
        Al.q[1] = *(const uint4*)(al + ao + 8);
        const int bo = ll * QSTR + c * 32 + 16 * half;  // B: q-col, 16 consec d
        Bh.q[0] = *(const uint4*)(qhP + bo);
        Bh.q[1] = *(const uint4*)(qhP + bo + 8);
        Bl.q[0] = *(const uint4*)(qlP + bo);
        Bl.q[1] = *(const uint4*)(qlP + bo + 8);
        acc = wmma_bf16(Ah.v, Bh.v, acc);   // Kh*Qh
        acc = wmma_bf16(Al.v, Bh.v, acc);   // Kl*Qh
        acc = wmma_bf16(Ah.v, Bl.v, acc);   // Kh*Ql
      }
#pragma unroll
      for (int v = 0; v < 8; ++v)
        S[(k0 + v + 8 * half) * SSTR + ll] = acc[v];   // lane-contiguous rows
    }
  }
  __syncthreads();

  // ---- softmax over k (rows of S^T); keep exp unnormalized ----
  {
    const int q = tid >> 4, seg = tid & 15;   // 16 threads per q, same half-wave
    float m = -3.402823466e38f;
#pragma unroll 8
    for (int i = 0; i < 128; ++i)
      m = fmaxf(m, S[(seg + 16 * i) * SSTR + q]);
#pragma unroll
    for (int off = 1; off < 16; off <<= 1)
      m = fmaxf(m, __shfl_xor(m, off, 32));
    float sum = 0.f;
#pragma unroll 8
    for (int i = 0; i < 128; ++i) {
      const int idx = (seg + 16 * i) * SSTR + q;
      float e = __expf(S[idx] - m);
      sum += e;
      S[idx] = e;
    }
#pragma unroll
    for (int off = 1; off < 16; off <<= 1)
      sum += __shfl_xor(sum, off, 32);
    if (seg == 0) sInv[q] = 1.0f / sum;
  }
  __syncthreads();

  // ---- P^T -> bf16 hi/lo planes [q][k] (overlays dead Q region) ----
  {
    const int q = tid & 15, seg = tid >> 4;
    uint32_t* ph = (uint32_t*)(smem + PHI_O);
    uint32_t* pl = (uint32_t*)(smem + PLO_O);
#pragma unroll 4
    for (int j = 0; j < 128; j += 2) {
      const int k = seg * 128 + j;
      float e0 = S[k * SSTR + q];
      float e1 = S[(k + 1) * SSTR + q];
      uint32_t h, l;
      splitT(e0, e1, h, l);
      const int di = (q * PSTR + k) >> 1;
      ph[di] = h; pl[di] = l;
    }
  }
  __syncthreads();

  // ---- pass 2: ctx^T = V^T * P^T  (A = Vt plane, B = P^T from LDS) ----
  {
    const uint16_t* vtb = vt + (size_t)b * Dn * TKn;
    const uint16_t* phP = (const uint16_t*)(smem + PHI_O);
    const uint16_t* plP = (const uint16_t*)(smem + PLO_O);
    f32x8 acc[8];
#pragma unroll
    for (int dt = 0; dt < 8; ++dt)
      acc[dt] = (f32x8){0.f, 0.f, 0.f, 0.f, 0.f, 0.f, 0.f, 0.f};

    for (int c2 = 0; c2 < TKn / 32; ++c2) {
      const int kb = c2 * 32;
      BF16Vec Bh, Bl;                         // B: q-col, 16 consec k (shared)
      const int bo = ll * PSTR + kb + 16 * half;
      Bh.q[0] = *(const uint4*)(phP + bo);
      Bh.q[1] = *(const uint4*)(phP + bo + 8);
      Bl.q[0] = *(const uint4*)(plP + bo);
      Bl.q[1] = *(const uint4*)(plP + bo + 8);
#pragma unroll
      for (int dt = 0; dt < 8; ++dt) {
        const int d = (w * 8 + dt) * 16 + ll;
        const uint16_t* ap = vtb + (size_t)d * TKn + kb + 16 * half;
        BF16Vec Av;                           // A: d-row, 16 consec k
        Av.q[0] = *(const uint4*)(ap);
        Av.q[1] = *(const uint4*)(ap + 8);
        acc[dt] = wmma_bf16(Av.v, Bh.v, acc[dt]);   // V*Ph
        acc[dt] = wmma_bf16(Av.v, Bl.v, acc[dt]);   // V*Pl
      }
    }

    const float inv = sInv[ll];               // column q = ll for all vgprs
    float* ob = out + ((size_t)b * TQn + q0 + ll) * Dn;
#pragma unroll
    for (int dt = 0; dt < 8; ++dt) {
      const int d0 = (w * 8 + dt) * 16;
#pragma unroll
      for (int v = 0; v < 8; ++v)
        ob[d0 + v + 8 * half] = acc[dt][v] * inv;
    }
  }
}

// ---------------------------------------------------------------------------
// Fallback: round-1 kernel with inline conversions (no workspace needed)
// ---------------------------------------------------------------------------
constexpr int FB_QHI = 0, FB_QLO = 32 * 1024, FB_S = 64 * 1024;
constexpr int FB_INV = FB_S + QT * TKn * 4;
constexpr int FB_LDS = FB_INV + 64;

__global__ __launch_bounds__(256) void attn_fallback(
    const float* __restrict__ enc, const float* __restrict__ dec,
    float* __restrict__ out) {
  extern __shared__ char smem[];
  unsigned short* qhi = (unsigned short*)(smem + FB_QHI);
  unsigned short* qlo = (unsigned short*)(smem + FB_QLO);
  float* S = (float*)(smem + FB_S);
  float* sInv = (float*)(smem + FB_INV);
  const int tid = threadIdx.x, b = blockIdx.y, q0 = blockIdx.x * QT;
  const int w = tid >> 5, lane = tid & 31, ll = lane & 15, half = lane >> 4;
  const float* Kb = enc + (size_t)b * TKn * Dn;
  const float* Qb = dec + (size_t)b * TQn * Dn;
  {
    const int row = tid >> 4, seg = tid & 15;
    const float4* src = (const float4*)(Qb + (size_t)(q0 + row) * Dn + seg * 64);
    uint32_t* qh = (uint32_t*)qhi;
    uint32_t* ql = (uint32_t*)qlo;
#pragma unroll
    for (int i = 0; i < 16; ++i) {
      float4 f = src[i];
      uint32_t h0, h1, l0, l1;
      splitT(f.x, f.y, h0, l0);
      splitT(f.z, f.w, h1, l1);
      int di = row * 512 + seg * 32 + i * 2;
      qh[di] = h0; qh[di + 1] = h1;
      ql[di] = l0; ql[di + 1] = l1;
    }
  }
  __syncthreads();
  for (int kt = w; kt < TKn / 16; kt += NWAVE) {
    const int k0 = kt * 16;
    f32x8 acc = {0.f, 0.f, 0.f, 0.f, 0.f, 0.f, 0.f, 0.f};
    const float* kp = Kb + (size_t)(k0 + ll) * Dn + 16 * half;
#pragma unroll 2
    for (int c = 0; c < Dn / 32; ++c) {
      BF16Vec ahi, alo;
      const int qoff = ll * Dn + c * 32 + 16 * half;
      ahi.q[0] = *(const uint4*)(qhi + qoff);
      ahi.q[1] = *(const uint4*)(qhi + qoff + 8);
      alo.q[0] = *(const uint4*)(qlo + qoff);
      alo.q[1] = *(const uint4*)(qlo + qoff + 8);
      const float4* kp4 = (const float4*)(kp + c * 32);
      BF16Vec bhi, blo;
#pragma unroll
      for (int j = 0; j < 4; ++j) {
        float4 f = kp4[j];
        splitT(f.x, f.y, bhi.u[2 * j], blo.u[2 * j]);
        splitT(f.z, f.w, bhi.u[2 * j + 1], blo.u[2 * j + 1]);
      }
      acc = wmma_bf16(ahi.v, bhi.v, acc);
      acc = wmma_bf16(alo.v, bhi.v, acc);
      acc = wmma_bf16(ahi.v, blo.v, acc);
    }
#pragma unroll
    for (int v = 0; v < 8; ++v) S[(v + 8 * half) * TKn + k0 + ll] = acc[v];
  }
  __syncthreads();
  {
    const int row = tid >> 4, seg = tid & 15;
    float4* S4 = (float4*)(S + row * TKn) + seg * 32;
    float m = -3.402823466e38f;
#pragma unroll 4
    for (int i = 0; i < 32; ++i) {
      float4 s = S4[i];
      m = fmaxf(m, fmaxf(fmaxf(s.x, s.y), fmaxf(s.z, s.w)));
    }
#pragma unroll
    for (int off = 1; off < 16; off <<= 1) m = fmaxf(m, __shfl_xor(m, off, 32));
    float sum = 0.f;
#pragma unroll 4
    for (int i = 0; i < 32; ++i) {
      float4 s = S4[i];
      s.x = __expf(s.x - m); s.y = __expf(s.y - m);
      s.z = __expf(s.z - m); s.w = __expf(s.w - m);
      sum += s.x + s.y + s.z + s.w;
      S4[i] = s;
    }
#pragma unroll
    for (int off = 1; off < 16; off <<= 1) sum += __shfl_xor(sum, off, 32);
    if (seg == 0) sInv[row] = 1.0f / sum;
  }
  __syncthreads();
  {
    f32x8 acc[8];
#pragma unroll
    for (int dt = 0; dt < 8; ++dt)
      acc[dt] = (f32x8){0.f, 0.f, 0.f, 0.f, 0.f, 0.f, 0.f, 0.f};
    for (int c2 = 0; c2 < TKn / 32; ++c2) {
      BF16Vec ahi, alo;
      const float* pr = S + ll * TKn + c2 * 32 + 8 * half;
      float4 p0 = *(const float4*)(pr);
      float4 p1 = *(const float4*)(pr + 4);
      float4 p2 = *(const float4*)(pr + 16);
      float4 p3 = *(const float4*)(pr + 20);
      splitT(p0.x, p0.y, ahi.u[0], alo.u[0]);
      splitT(p0.z, p0.w, ahi.u[1], alo.u[1]);
      splitT(p1.x, p1.y, ahi.u[2], alo.u[2]);
      splitT(p1.z, p1.w, ahi.u[3], alo.u[3]);
      splitT(p2.x, p2.y, ahi.u[4], alo.u[4]);
      splitT(p2.z, p2.w, ahi.u[5], alo.u[5]);
      splitT(p3.x, p3.y, ahi.u[6], alo.u[6]);
      splitT(p3.z, p3.w, ahi.u[7], alo.u[7]);
      const float* vbase = Kb + (size_t)(c2 * 32 + 8 * half) * Dn + ll;
#pragma unroll
      for (int dt = 0; dt < 8; ++dt) {
        const float* vp = vbase + (w * 8 + dt) * 16;
        float vv[16];
#pragma unroll
        for (int j = 0; j < 16; ++j) vv[j] = vp[(j + (j & 8)) * Dn];
        BF16Vec bhi, blo;
#pragma unroll
        for (int j = 0; j < 8; ++j)
          splitT(vv[2 * j], vv[2 * j + 1], bhi.u[j], blo.u[j]);
        acc[dt] = wmma_bf16(ahi.v, bhi.v, acc[dt]);
        acc[dt] = wmma_bf16(alo.v, bhi.v, acc[dt]);
        acc[dt] = wmma_bf16(ahi.v, blo.v, acc[dt]);
      }
    }
    float inv[8];
#pragma unroll
    for (int v = 0; v < 8; ++v) inv[v] = sInv[v + 8 * half];
    float* ob = out + ((size_t)b * TQn + q0) * Dn;
#pragma unroll
    for (int dt = 0; dt < 8; ++dt) {
      const int d0 = (w * 8 + dt) * 16;
#pragma unroll
      for (int v = 0; v < 8; ++v)
        ob[(size_t)(v + 8 * half) * Dn + d0 + ll] = acc[dt][v] * inv[v];
    }
  }
}

extern "C" void kernel_launch(void* const* d_in, const int* in_sizes, int n_in,
                              void* d_out, int out_size, void* d_ws,
                              size_t ws_size, hipStream_t stream) {
  (void)in_sizes; (void)n_in; (void)out_size;
  const float* enc = (const float*)d_in[0];  // encoder_hiddens [B,Tk,D]
  const float* dec = (const float*)d_in[1];  // decoder_hidden  [B,Tq,D]
  float* out = (float*)d_out;

  if (ws_size >= WS_NEED && d_ws != nullptr) {
    uint16_t* khi = (uint16_t*)((char*)d_ws + KHI_OFF);
    uint16_t* klo = (uint16_t*)((char*)d_ws + KLO_OFF);
    uint16_t* vtp = (uint16_t*)((char*)d_ws + VT_OFF);
    const int nelem = Bn * TKn * Dn;                       // 33554432
    conv_k_planes<<<nelem / 8 / 256, 256, 0, stream>>>(enc, khi, klo);
    conv_vt_plane<<<dim3(TKn / 64, Dn / 64, Bn), 256, 0, stream>>>(enc, vtp);
    attn_ws<<<dim3(TQn / QT, Bn), 256, LDSZ, stream>>>(dec, khi, klo, vtp, out);
  } else {
    attn_fallback<<<dim3(TQn / QT, Bn), 256, FB_LDS, stream>>>(enc, dec, out);
  }
}
